// SimplePatchGNN_37709812859357
// MI455X (gfx1250) — matrix operand order repaired and checked by reading, not compiled, not run on hardware
//
#include <hip/hip_runtime.h>
#include <hip/hip_bf16.h>
#include <stdint.h>

// ---------------------------------------------------------------------------
// Types for WMMA
// ---------------------------------------------------------------------------
typedef __attribute__((ext_vector_type(16))) _Float16 v16h;
typedef __attribute__((ext_vector_type(8)))  float    v8f;

union Frag {
    v16h     v;
    _Float16 h[16];
    uint32_t u[8];
};

#define HID_D   512
#define NVAR    256
#define BDIM    8
#define LSEQ    512
#define LPRED   48
#define NHEAD   8
#define HD      64      // head dim
#define RH      32      // rh
#define RFD     64      // r = 2*rh
#define KPOOL   8
#define PCH     16
#define TED     10
#define TOKENS  (BDIM * NVAR)          // 2048
#define DECROWS (BDIM * NVAR * LPRED)  // 98304

// ---------------------------------------------------------------------------
// Generic WMMA GEMM:  C[M,N] = A[M,K](f16) * Bw[N,K]^T(f16)  (+bias)(+resid)(relu)
// One wave computes a 16x64 macro-tile (4 accumulators, A fragment reused 4x).
// M mult of 16, N mult of 64, K mult of 32.
// ---------------------------------------------------------------------------
__global__ void k_gemm(const _Float16* __restrict__ A, const _Float16* __restrict__ Bw,
                       float* __restrict__ C, const float* __restrict__ bias,
                       const float* __restrict__ resid, int M, int Nn, int K, int do_relu)
{
    int wave = threadIdx.x >> 5;
    int lane = threadIdx.x & 31;
    int ngrp = Nn >> 6;                         // groups of 64 columns
    int tile = blockIdx.x * (blockDim.x >> 5) + wave;
    int total = (M >> 4) * ngrp;
    if (tile >= total) return;                  // wave-uniform
    int tm = tile / ngrp, tg = tile % ngrp;
    int mbase = tm << 4, nbase = tg << 6;
    int m16 = lane & 15, kh = lane >> 4;

    const uint32_t* Ap  = (const uint32_t*)(A + (size_t)(mbase + m16) * K);
    const uint32_t* Bp0 = (const uint32_t*)(Bw + (size_t)(nbase + m16) * K);
    size_t rstride = (size_t)(K >> 1) * 16;     // dwords per 16 B-rows

    v8f acc[4] = {{}, {}, {}, {}};
    for (int k = 0; k < K; k += 32) {
        int base = k >> 1;                      // dword index of k
        Frag fa;
        #pragma unroll
        for (int i = 0; i < 4; ++i) {
            // A 16x32 f16 layout: VGPR i<4: K=8*kh+2i ; i>=4: K=16+8*kh+2(i-4)
            fa.u[i]     = Ap[base + 4*kh + i];
            fa.u[i + 4] = Ap[base + 8 + 4*kh + i];
        }
        if (k + 32 < K) {
            // near-cache (WGP-scope) prefetch of next k-step
            __builtin_prefetch((const void*)(Ap + base + 16), 0, 3);
            __builtin_prefetch((const void*)(Bp0 + base + 16), 0, 3);
        }
        #pragma unroll
        for (int s = 0; s < 4; ++s) {
            Frag fb;
            const uint32_t* Bp = Bp0 + rstride * s;
            #pragma unroll
            for (int i = 0; i < 8; ++i) {
                // B 32x16 f16 layout: VGPR i<4: K=8*kh+2i ; i>=4: K=16+8*kh+2(i-4)
                fb.u[i] = Bp[base + 4*kh + (i & 3) + 8*(i >> 2)];
            }
            acc[s] = __builtin_amdgcn_wmma_f32_16x16x32_f16(false, fa.v, false, fb.v,
                                                            (short)0, acc[s], false, false);
        }
    }

    #pragma unroll
    for (int s = 0; s < 4; ++s) {
        int col = nbase + 16*s + m16;
        float bv = bias ? bias[col] : 0.0f;
        #pragma unroll
        for (int r = 0; r < 8; ++r) {
            int row = mbase + 8*kh + r;
            size_t idx = (size_t)row * Nn + col;
            float v = acc[s][r] + bv;
            if (resid)   v += resid[idx];
            if (do_relu) v = v > 0.f ? v : 0.f;
            C[idx] = v;
        }
    }
}

// ---------------------------------------------------------------------------
// Fused decoder: A = relu(d1z[b,n,:] + teterm[b,lp,:]) built on the fly,
// 16x64 macro-tile GEMM with dec2 weights, epilogue fuses relu + dec3 dot +
// cross-lane reduce + atomicAdd into out (B, LP, N).  rows ordered (b, n, lp).
// ---------------------------------------------------------------------------
__global__ void k_dec2_fused(const float* __restrict__ d1z, const float* __restrict__ teterm,
                             const _Float16* __restrict__ Bw, const float* __restrict__ bias2,
                             const float* __restrict__ w3, float* __restrict__ out)
{
    const int K = HID_D, Nn = HID_D;
    int wave = threadIdx.x >> 5;
    int lane = threadIdx.x & 31;
    int ngrp = Nn >> 6;                          // 8 groups of 64 cols
    int tile = blockIdx.x * (blockDim.x >> 5) + wave;
    int total = (DECROWS >> 4) * ngrp;
    if (tile >= total) return;
    int tm = tile / ngrp, tg = tile % ngrp;
    int mbase = tm << 4, nbase = tg << 6;
    int m16 = lane & 15, kh = lane >> 4;

    int rowA = mbase + m16;
    int lp = rowA % LPRED; int t1 = rowA / LPRED;
    int nv = t1 % NVAR;    int bb = t1 / NVAR;
    const float* za = d1z    + (size_t)(bb * NVAR  + nv) * K;
    const float* ta = teterm + (size_t)(bb * LPRED + lp) * K;
    const uint32_t* Bp0 = (const uint32_t*)(Bw + (size_t)(nbase + m16) * K);
    size_t rstride = (size_t)(K >> 1) * 16;

    v8f acc[4] = {{}, {}, {}, {}};
    for (int k = 0; k < K; k += 32) {
        Frag fa;
        #pragma unroll
        for (int e = 0; e < 16; ++e) {
            int i = e >> 1, w = e & 1;
            int ko = (i < 4) ? (8*kh + 2*i) : (16 + 8*kh + 2*(i - 4));
            int kk = k + ko + w;
            float vv = za[kk] + ta[kk];
            fa.h[e] = (_Float16)(vv > 0.f ? vv : 0.f);
        }
        int base = k >> 1;
        if (k + 32 < K) {
            __builtin_prefetch((const void*)(za + k + 32), 0, 3);
            __builtin_prefetch((const void*)(Bp0 + base + 16), 0, 3);
        }
        #pragma unroll
        for (int s = 0; s < 4; ++s) {
            Frag fb;
            const uint32_t* Bp = Bp0 + rstride * s;
            #pragma unroll
            for (int i = 0; i < 8; ++i) fb.u[i] = Bp[base + 4*kh + (i & 3) + 8*(i >> 2)];
            acc[s] = __builtin_amdgcn_wmma_f32_16x16x32_f16(false, fa.v, false, fb.v,
                                                            (short)0, acc[s], false, false);
        }
    }

    float bc[4], wc[4];
    #pragma unroll
    for (int s = 0; s < 4; ++s) {
        int col = nbase + 16*s + m16;
        bc[s] = bias2[col];
        wc[s] = w3[col];
    }
    #pragma unroll
    for (int r = 0; r < 8; ++r) {
        float p = 0.f;
        #pragma unroll
        for (int s = 0; s < 4; ++s) {
            float v = acc[s][r] + bc[s];
            v = v > 0.f ? v : 0.f;
            p += v * wc[s];
        }
        p += __shfl_xor(p, 1);
        p += __shfl_xor(p, 2);
        p += __shfl_xor(p, 4);
        p += __shfl_xor(p, 8);
        if (m16 == 0) {
            int row = mbase + 8*kh + r;
            int lp2 = row % LPRED; int t2 = row / LPRED;
            int n2  = t2 % NVAR;   int b2i = t2 / NVAR;
            atomicAdd(out + ((size_t)(b2i * LPRED + lp2) * NVAR + n2), p);
        }
    }
}

// ---------------------------------------------------------------------------
// Build rfft/irfft 512x512 matrices (f16) for the channel-axis DFT.
// fwd[j,l]: j<=256 -> cos(2*pi*j*l/512)/512 ; j>=257 -> sin(2*pi*(j-257)*l/512)/512
// inv[l,j]: j==0 -> 1 ; 1<=j<=255 -> 2cos ; j==256 -> 0 ; j>=257 -> 2sin(j-256)
// ---------------------------------------------------------------------------
__global__ void k_fftmats(_Float16* __restrict__ fwd, _Float16* __restrict__ inv)
{
    int i = blockIdx.x * blockDim.x + threadIdx.x;
    if (i >= HID_D * HID_D) return;
    int r = i >> 9, c = i & 511;
    const float TWO_PI = 6.283185307179586f;
    float fv;
    if (r <= 256) fv = cosf(TWO_PI * (float)r * (float)c / 512.f) / 512.f;
    else          fv = sinf(TWO_PI * (float)(r - 257) * (float)c / 512.f) / 512.f;
    fwd[i] = (_Float16)fv;
    float iv;
    if (c == 0)        iv = 1.f;
    else if (c < 256)  iv = 2.f * cosf(TWO_PI * (float)c * (float)r / 512.f);
    else if (c == 256) iv = 0.f;
    else               iv = 2.f * sinf(TWO_PI * (float)(c - 256) * (float)r / 512.f);
    inv[i] = (_Float16)iv;
}

// ---------------------------------------------------------------------------
// f32 -> f16 conversions
// ---------------------------------------------------------------------------
__global__ void k_cvt(const float* __restrict__ s, _Float16* __restrict__ d, int n)
{
    int i = blockIdx.x * blockDim.x + threadIdx.x;
    if (i < n) d[i] = (_Float16)s[i];
}
__global__ void k_cvt_strided(const float* __restrict__ s, _Float16* __restrict__ d,
                              int rows, int cols, int stride)
{
    int i = blockIdx.x * blockDim.x + threadIdx.x;
    if (i >= rows * cols) return;
    int r = i / cols, c = i % cols;
    d[i] = (_Float16)s[(size_t)r * stride + c];
}

__global__ void k_fill_out(float* __restrict__ out, const float* __restrict__ b3, int n)
{
    int i = blockIdx.x * blockDim.x + threadIdx.x;
    if (i < n) out[i] = b3[0];
}

// ---------------------------------------------------------------------------
// Stage 1+2: conv preprocessing + te-embed + Gaussian kernel pooling + feat_proj
// One block per (b, n).  Output z[token=b*N+n, 512] including positional enc.
// ---------------------------------------------------------------------------
__global__ void k_preproc(const float* __restrict__ X, const float* __restrict__ mask,
                          const float* __restrict__ tp_true,
                          const float* __restrict__ cw1, const float* __restrict__ cb1,
                          const float* __restrict__ cw2, const float* __restrict__ cb2,
                          const float* __restrict__ tsw, const float* __restrict__ tsb,
                          const float* __restrict__ tsinw, const float* __restrict__ tsinb,
                          const float* __restrict__ tcosw, const float* __restrict__ tcosb,
                          const float* __restrict__ tpw, const float* __restrict__ tpb,
                          const float* __restrict__ intra_c, const float* __restrict__ intra_la,
                          const float* __restrict__ intra_gate,
                          const float* __restrict__ fpw, const float* __restrict__ fpb,
                          float* __restrict__ z)
{
    int b = blockIdx.x >> 8;
    int n = blockIdx.x & 255;
    int tid = threadIdx.x;

    __shared__ float sX[LSEQ], sM[LSEQ], sXf[LSEQ];
    __shared__ float sSw[KPOOL], sSwx[KPOOL], sSm;
    __shared__ float sC[KPOOL], sA[KPOOL], sG[KPOOL];

    if (tid < KPOOL) {
        sSw[tid] = 0.f; sSwx[tid] = 0.f;
        sC[tid] = intra_c[tid];
        sA[tid] = expf(intra_la[tid]) + 1e-6f;
        sG[tid] = 1.f / (1.f + expf(-intra_gate[tid]));
    }
    if (tid == 0) sSm = 0.f;
    for (int l = tid; l < LSEQ; l += blockDim.x) {
        size_t idx = ((size_t)b * LSEQ + l) * NVAR + n;
        sX[l] = X[idx];
        sM[l] = mask[idx];
    }
    __syncthreads();

    float tmin = tp_true[b * LSEQ];            // tp_true is sorted
    float tmax = tp_true[b * LSEQ + LSEQ - 1];

    for (int l = tid; l < LSEQ; l += blockDim.x) {
        float xm1 = (l > 0)       ? sX[l - 1] : 0.f;
        float x0  = sX[l];
        float xp1 = (l < LSEQ-1)  ? sX[l + 1] : 0.f;
        float acc = cb2[0];
        #pragma unroll
        for (int c = 0; c < PCH; ++c) {
            float h = cw1[c*3+0]*xm1 + cw1[c*3+1]*x0 + cw1[c*3+2]*xp1 + cb1[c];
            h = h > 0.f ? h : 0.f;
            acc += cw2[c] * h;
        }
        float t = tp_true[b * LSEQ + l];
        float s = tpb[0] + tpw[0] * (tsw[0]*t + tsb[0]);
        #pragma unroll
        for (int i = 0; i < 4; ++i) s += tpw[1+i] * sinf(tsinw[i]*t + tsinb[i]);
        #pragma unroll
        for (int i = 0; i < 5; ++i) s += tpw[5+i] * cosf(tcosw[i]*t + tcosb[i]);
        sXf[l] = acc + s;
    }
    __syncthreads();

    float psw[KPOOL], pswx[KPOOL], psm = 0.f;
    #pragma unroll
    for (int kk = 0; kk < KPOOL; ++kk) { psw[kk] = 0.f; pswx[kk] = 0.f; }
    for (int l = tid; l < LSEQ; l += blockDim.x) {
        float t  = tp_true[b * LSEQ + l];
        float tn = (t - tmin) / (tmax - tmin + 1e-8f);
        float m  = sM[l];  psm += m;
        float xf = sXf[l];
        #pragma unroll
        for (int kk = 0; kk < KPOOL; ++kk) {
            float td = tn - sC[kk];
            float w = expf(-0.5f * td * td / (sA[kk] * sA[kk])) * m;
            psw[kk] += w; pswx[kk] += w * xf;
        }
    }
    #pragma unroll
    for (int kk = 0; kk < KPOOL; ++kk) {
        atomicAdd(&sSw[kk], psw[kk]);
        atomicAdd(&sSwx[kk], pswx[kk]);
    }
    atomicAdd(&sSm, psm);
    __syncthreads();

    float flag = sSm > 0.f ? 1.f : 0.f;
    float hk[KPOOL];
    #pragma unroll
    for (int kk = 0; kk < KPOOL; ++kk) hk[kk] = sSwx[kk] / (sSw[kk] + 1e-8f) * sG[kk];

    for (int c = tid; c < HID_D; c += blockDim.x) {
        float acc = fpb[c];
        #pragma unroll
        for (int kk = 0; kk < KPOOL; ++kk) acc += fpw[c * 9 + kk] * hk[kk];
        acc += fpw[c * 9 + 8] * flag;
        int i2 = c >> 1;
        float div = expf(-(float)(2 * i2) * 9.210340371976184f / 512.f);
        float pe = (c & 1) ? cosf((float)n * div) : sinf((float)n * div);
        z[(size_t)blockIdx.x * HID_D + c] = acc + pe;
    }
}

// ---------------------------------------------------------------------------
// LayerNorm over 512, writes f16 for GEMM-A consumption.
// ---------------------------------------------------------------------------
__global__ void k_layernorm(const float* __restrict__ x, const float* __restrict__ g,
                            const float* __restrict__ be, _Float16* __restrict__ out)
{
    const int D = HID_D;
    int t = blockIdx.x, tid = threadIdx.x;
    const float* xr = x + (size_t)t * D;
    __shared__ float s1[256], s2[256];
    float a = 0.f, b = 0.f;
    for (int i = tid; i < D; i += 256) { float v = xr[i]; a += v; b += v * v; }
    s1[tid] = a; s2[tid] = b; __syncthreads();
    for (int o = 128; o > 0; o >>= 1) {
        if (tid < o) { s1[tid] += s1[tid + o]; s2[tid] += s2[tid + o]; }
        __syncthreads();
    }
    float mean = s1[0] / D;
    float var  = s2[0] / D - mean * mean;
    float rstd = rsqrtf(var + 1e-5f);
    for (int i = tid; i < D; i += 256)
        out[(size_t)t * D + i] = (_Float16)((xr[i] - mean) * rstd * g[i] + be[i]);
}

// ---------------------------------------------------------------------------
// Random Fourier features: phi[t,h,0:32]=cos(proj)/sqrt(32), [32:64]=sin(proj)/sqrt(32)
// proj = q[t, h*64:+64] . W[h,:,j] + bph[h,j]
// ---------------------------------------------------------------------------
__global__ void k_rff(const float* __restrict__ q, const float* __restrict__ W,
                      const float* __restrict__ bph, float* __restrict__ phi, int T)
{
    int gid = blockIdx.x * blockDim.x + threadIdx.x;
    int total = T * NHEAD * RH;
    if (gid >= total) return;
    int j = gid & 31;
    int h = (gid >> 5) & 7;
    int t = gid >> 8;
    const float* qv = q + (size_t)t * HID_D + h * HD;
    const float* Wv = W + (size_t)(h * HD) * RH + j;
    float p = bph[h * RH + j];
    #pragma unroll 8
    for (int d = 0; d < HD; ++d) p += qv[d] * Wv[d * RH];
    const float inv = 0.17677669529663687f;   // 1/sqrt(32)
    size_t base = ((size_t)t * NHEAD + h) * RFD;
    phi[base + j]      = cosf(p) * inv;
    phi[base + RH + j] = sinf(p) * inv;
}

// ---------------------------------------------------------------------------
// Linear attention: KV[b,h,r,d] = sum_m phi_k * v ; Ksum[b,h,r] = sum_m phi_k
// ---------------------------------------------------------------------------
__global__ void k_kv(const float* __restrict__ phik, const float* __restrict__ v,
                     float* __restrict__ KV, float* __restrict__ Ksum)
{
    int bh = blockIdx.x;                // 0..63
    int b = bh >> 3, h = bh & 7;
    for (int idx = threadIdx.x; idx < RFD * HD; idx += blockDim.x) {
        int r = idx >> 6, d = idx & 63;
        float acc = 0.f;
        for (int m = 0; m < NVAR; ++m) {
            size_t t = (size_t)b * NVAR + m;
            acc += phik[(t * NHEAD + h) * RFD + r] * v[t * HID_D + h * HD + d];
        }
        KV[(size_t)bh * (RFD * HD) + idx] = acc;
    }
    for (int r = threadIdx.x; r < RFD; r += blockDim.x) {
        float acc = 0.f;
        for (int m = 0; m < NVAR; ++m)
            acc += phik[(((size_t)b * NVAR + m) * NHEAD + h) * RFD + r];
        Ksum[bh * RFD + r] = acc;
    }
}

// num/den -> merged heads out[t, 512]
__global__ void k_attnout(const float* __restrict__ phiq, const float* __restrict__ KV,
                          const float* __restrict__ Ksum, float* __restrict__ out)
{
    int t = blockIdx.x;
    int b = t / NVAR;
    __shared__ float den[NHEAD];
    if (threadIdx.x < NHEAD) {
        int h = threadIdx.x;
        float a = 0.f;
        const float* pq = phiq + ((size_t)t * NHEAD + h) * RFD;
        const float* ks = Ksum + (b * NHEAD + h) * RFD;
        #pragma unroll 8
        for (int r = 0; r < RFD; ++r) a += pq[r] * ks[r];
        den[h] = a + 1e-6f;
    }
    __syncthreads();
    for (int idx = threadIdx.x; idx < HID_D; idx += blockDim.x) {
        int h = idx >> 6, d = idx & 63;
        const float* pq = phiq + ((size_t)t * NHEAD + h) * RFD;
        const float* kv = KV + ((size_t)(b * NHEAD + h)) * (RFD * HD) + d;
        float num = 0.f;
        #pragma unroll 8
        for (int r = 0; r < RFD; ++r) num += pq[r] * kv[r * HD];
        out[(size_t)t * HID_D + idx] = num / den[h];
    }
}

// ---------------------------------------------------------------------------
// Decoder te-part: teterm[b,lp,c] = sum_e te(tp_pred[b,lp])_e * dec1.w[c,512+e] + dec1.b[c]
// ---------------------------------------------------------------------------
__global__ void k_teterm(const float* __restrict__ tp_pred,
                         const float* __restrict__ w1, const float* __restrict__ b1,
                         const float* __restrict__ tsw, const float* __restrict__ tsb,
                         const float* __restrict__ tsinw, const float* __restrict__ tsinb,
                         const float* __restrict__ tcosw, const float* __restrict__ tcosb,
                         float* __restrict__ teterm)
{
    int blp = blockIdx.x;                    // b*LP + lp
    float t = tp_pred[blp];
    __shared__ float te[TED];
    if (threadIdx.x == 0) {
        te[0] = tsw[0] * t + tsb[0];
        for (int i = 0; i < 4; ++i) te[1 + i] = sinf(tsinw[i] * t + tsinb[i]);
        for (int i = 0; i < 5; ++i) te[5 + i] = cosf(tcosw[i] * t + tcosb[i]);
    }
    __syncthreads();
    for (int c = threadIdx.x; c < HID_D; c += blockDim.x) {
        float a = b1[c];
        const float* wr = w1 + (size_t)c * (HID_D + TED) + HID_D;
        #pragma unroll
        for (int e = 0; e < TED; ++e) a += te[e] * wr[e];
        teterm[(size_t)blp * HID_D + c] = a;
    }
}

// ---------------------------------------------------------------------------
// Host orchestration
// ---------------------------------------------------------------------------
// input leaf indices (setup_inputs dict insertion order, params flattened recursively)
#define I_TP_PRED 0
#define I_X       1
#define I_TP_TRUE 2
#define I_MASK    3
#define I_PCW1 4
#define I_PCB1 5
#define I_PCW2 6
#define I_PCB2 7
#define I_TSW  8
#define I_TSB  9
#define I_TSINW 10
#define I_TSINB 11
#define I_TCOSW 12
#define I_TCOSB 13
#define I_TPW  14
#define I_TPB  15
#define I_INTRA_C 16
#define I_INTRA_LA 17
#define I_INTRA_G 18
#define I_FPW 19
#define I_FPB 20
#define I_VAW 21
#define I_VAB 22
#define I_D1W 23
#define I_D1B 24
#define I_D2W 25
#define I_D2B 26
#define I_D3W 27
#define I_D3B 28
#define I_BLK0 29
// per-block leaf offsets (18 leaves / block)
#define BO_N1G 0
#define BO_N1B 1
#define BO_N2G 2
#define BO_N2B 3
#define BO_QW  4
#define BO_QB  5
#define BO_KW  6
#define BO_KB  7
#define BO_VW  8
#define BO_VB  9
#define BO_PW  10
#define BO_PB  11
#define BO_W   12
#define BO_BPH 13
#define BO_M1W 14
#define BO_M1B 15
#define BO_M2W 16
#define BO_M2B 17

extern "C" void kernel_launch(void* const* d_in, const int* in_sizes, int n_in,
                              void* d_out, int out_size, void* d_ws, size_t ws_size,
                              hipStream_t stream)
{
    (void)in_sizes; (void)n_in; (void)out_size; (void)ws_size;
    auto IN = [&](int i) -> const float* { return (const float*)d_in[i]; };
    float* outp = (float*)d_out;

    // ---- workspace bump allocator ----
    char* cur = (char*)d_ws;
    auto alloc = [&](size_t bytes) -> void* {
        void* p = (void*)cur;
        cur += (bytes + 255) & ~(size_t)255;
        return p;
    };
    // f16 weight / matrix buffers
    _Float16* fftF   = (_Float16*)alloc((size_t)HID_D * HID_D * 2);
    _Float16* fftI   = (_Float16*)alloc((size_t)HID_D * HID_D * 2);
    _Float16* vagg16 = (_Float16*)alloc((size_t)HID_D * HID_D * 2);
    _Float16* d1w16  = (_Float16*)alloc((size_t)HID_D * HID_D * 2);
    _Float16* d2w16  = (_Float16*)alloc((size_t)HID_D * HID_D * 2);
    _Float16* qw16   = (_Float16*)alloc((size_t)HID_D * HID_D * 2);
    _Float16* kw16   = (_Float16*)alloc((size_t)HID_D * HID_D * 2);
    _Float16* vw16   = (_Float16*)alloc((size_t)HID_D * HID_D * 2);
    _Float16* pw16   = (_Float16*)alloc((size_t)HID_D * HID_D * 2);
    _Float16* m1w16  = (_Float16*)alloc((size_t)4 * HID_D * HID_D * 2);
    _Float16* m2w16  = (_Float16*)alloc((size_t)4 * HID_D * HID_D * 2);
    // activations
    float*    z      = (float*)alloc((size_t)TOKENS * HID_D * 4);
    _Float16* ln16   = (_Float16*)alloc((size_t)TOKENS * HID_D * 2);
    float*    fx32   = (float*)alloc((size_t)TOKENS * HID_D * 4);   // these four are
    float*    q32    = (float*)alloc((size_t)TOKENS * HID_D * 4);   // contiguous and
    float*    k32    = (float*)alloc((size_t)TOKENS * HID_D * 4);   // aliased by h32
    float*    v32    = (float*)alloc((size_t)TOKENS * HID_D * 4);   // (16 MB region)
    float*    h32    = fx32;                                        // [2048 x 2048] f32
    _Float16* h16    = (_Float16*)alloc((size_t)TOKENS * 4 * HID_D * 2);
    _Float16* fx16   = (_Float16*)alloc((size_t)TOKENS * HID_D * 2);
    float*    mid32  = (float*)alloc((size_t)TOKENS * HID_D * 4);
    _Float16* mid16  = (_Float16*)alloc((size_t)TOKENS * HID_D * 2);
    float*    proj32 = (float*)alloc((size_t)TOKENS * HID_D * 4);
    _Float16* proj16 = (_Float16*)alloc((size_t)TOKENS * HID_D * 2);
    float*    phiq   = (float*)alloc((size_t)TOKENS * NHEAD * RFD * 4);
    float*    phik   = (float*)alloc((size_t)TOKENS * NHEAD * RFD * 4);
    float*    KV     = (float*)alloc((size_t)BDIM * NHEAD * RFD * HD * 4);
    float*    Ksum   = (float*)alloc((size_t)BDIM * NHEAD * RFD * 4);
    float*    zagg32 = (float*)alloc((size_t)TOKENS * HID_D * 4);
    _Float16* zagg16 = (_Float16*)alloc((size_t)TOKENS * HID_D * 2);
    float*    d1z    = (float*)alloc((size_t)TOKENS * HID_D * 4);
    float*    teterm = (float*)alloc((size_t)BDIM * LPRED * HID_D * 4);

    auto cvt = [&](const float* s, _Float16* d, int n) {
        k_cvt<<<(n + 255) / 256, 256, 0, stream>>>(s, d, n);
    };
    auto gemm = [&](const _Float16* A, const _Float16* Bw, float* C, const float* bias,
                    const float* resid, int M, int Nn, int K, int relu) {
        int tiles = (M / 16) * (Nn / 64);       // 16x64 macro-tiles
        int blocks = (tiles + 3) / 4;
        k_gemm<<<blocks, 128, 0, stream>>>(A, Bw, C, bias, resid, M, Nn, K, relu);
    };

    // ---- one-time (per call) matrix/weight preparation ----
    k_fftmats<<<(HID_D * HID_D + 255) / 256, 256, 0, stream>>>(fftF, fftI);
    cvt(IN(I_VAW), vagg16, HID_D * HID_D);
    k_cvt_strided<<<(HID_D * HID_D + 255) / 256, 256, 0, stream>>>(
        IN(I_D1W), d1w16, HID_D, HID_D, HID_D + TED);
    cvt(IN(I_D2W), d2w16, HID_D * HID_D);

    // ---- stage 1+2: preprocessing -> z ----
    k_preproc<<<TOKENS, 256, 0, stream>>>(
        IN(I_X), IN(I_MASK), IN(I_TP_TRUE),
        IN(I_PCW1), IN(I_PCB1), IN(I_PCW2), IN(I_PCB2),
        IN(I_TSW), IN(I_TSB), IN(I_TSINW), IN(I_TSINB), IN(I_TCOSW), IN(I_TCOSB),
        IN(I_TPW), IN(I_TPB),
        IN(I_INTRA_C), IN(I_INTRA_LA), IN(I_INTRA_G),
        IN(I_FPW), IN(I_FPB), z);

    // ---- transformer blocks ----
    for (int blk = 0; blk < 4; ++blk) {
        int base = I_BLK0 + 18 * blk;
        const float* n1g = IN(base + BO_N1G); const float* n1b = IN(base + BO_N1B);
        const float* n2g = IN(base + BO_N2G); const float* n2b = IN(base + BO_N2B);
        cvt(IN(base + BO_QW),  qw16,  HID_D * HID_D);
        cvt(IN(base + BO_KW),  kw16,  HID_D * HID_D);
        cvt(IN(base + BO_VW),  vw16,  HID_D * HID_D);
        cvt(IN(base + BO_PW),  pw16,  HID_D * HID_D);
        cvt(IN(base + BO_M1W), m1w16, 4 * HID_D * HID_D);
        cvt(IN(base + BO_M2W), m2w16, 4 * HID_D * HID_D);

        // attention (pre-norm)
        k_layernorm<<<TOKENS, 256, 0, stream>>>(z, n1g, n1b, ln16);
        gemm(ln16, fftF, fx32, nullptr, nullptr, TOKENS, HID_D, HID_D, 0);   // rfft
        cvt(fx32, fx16, TOKENS * HID_D);
        gemm(fx16, qw16, q32, IN(base + BO_QB), nullptr, TOKENS, HID_D, HID_D, 0);
        gemm(fx16, kw16, k32, IN(base + BO_KB), nullptr, TOKENS, HID_D, HID_D, 0);
        gemm(fx16, vw16, v32, IN(base + BO_VB), nullptr, TOKENS, HID_D, HID_D, 0);
        int rffT = TOKENS * NHEAD * RH;
        k_rff<<<(rffT + 255) / 256, 256, 0, stream>>>(q32, IN(base + BO_W), IN(base + BO_BPH), phiq, TOKENS);
        k_rff<<<(rffT + 255) / 256, 256, 0, stream>>>(k32, IN(base + BO_W), IN(base + BO_BPH), phik, TOKENS);
        k_kv<<<BDIM * NHEAD, 256, 0, stream>>>(phik, v32, KV, Ksum);
        k_attnout<<<TOKENS, 256, 0, stream>>>(phiq, KV, Ksum, mid32);
        cvt(mid32, mid16, TOKENS * HID_D);
        gemm(mid16, pw16, proj32, IN(base + BO_PB), nullptr, TOKENS, HID_D, HID_D, 0);
        cvt(proj32, proj16, TOKENS * HID_D);
        gemm(proj16, fftI, z, nullptr, z, TOKENS, HID_D, HID_D, 0);          // irfft + residual

        // MLP (pre-norm)
        k_layernorm<<<TOKENS, 256, 0, stream>>>(z, n2g, n2b, ln16);
        gemm(ln16, m1w16, h32, IN(base + BO_M1B), nullptr, TOKENS, 4 * HID_D, HID_D, 1);
        cvt(h32, h16, TOKENS * 4 * HID_D);
        gemm(h16, m2w16, z, IN(base + BO_M2B), z, TOKENS, HID_D, 4 * HID_D, 0);
    }

    // ---- var_agg + decoder ----
    cvt(z, ln16, TOKENS * HID_D);
    gemm(ln16, vagg16, zagg32, IN(I_VAB), nullptr, TOKENS, HID_D, HID_D, 0);
    cvt(zagg32, zagg16, TOKENS * HID_D);
    gemm(zagg16, d1w16, d1z, nullptr, nullptr, TOKENS, HID_D, HID_D, 0);     // dec1 z-part
    k_teterm<<<BDIM * LPRED, 256, 0, stream>>>(
        IN(I_TP_PRED), IN(I_D1W), IN(I_D1B),
        IN(I_TSW), IN(I_TSB), IN(I_TSINW), IN(I_TSINB), IN(I_TCOSW), IN(I_TCOSB),
        teterm);
    k_fill_out<<<(DECROWS + 255) / 256, 256, 0, stream>>>(outp, IN(I_D3B), DECROWS);
    {
        int tiles = (DECROWS / 16) * (HID_D / 64);
        int blocks = (tiles + 3) / 4;
        k_dec2_fused<<<blocks, 128, 0, stream>>>(d1z, teterm, d2w16,
                                                 IN(I_D2B), IN(I_D3W), outp);
    }
}
